// lstm_50345606643785
// MI455X (gfx1250) — matrix operand order repaired and checked by reading, not compiled, not run on hardware
//
#include <hip/hip_runtime.h>
#include <hip/hip_bf16.h>
#include <math.h>
#include <stdint.h>

#define B_ 64
#define R_ 196
#define F_ 512
#define H_ 512
#define V_ 32000
#define T_ 30
#define START_TOK 2

typedef __bf16 bf16_t;
typedef __bf16 v16bf __attribute__((ext_vector_type(16)));
typedef __bf16 v8bf  __attribute__((ext_vector_type(8)));
typedef float  v8f   __attribute__((ext_vector_type(8)));

// ---------------------------------------------------------------------------
// bf16 WMMA GEMM:  C[M,N] (fp32, ld ldc) = A[M,K] (ld lda) @ W[N,K]^T (ld ldw)
// + bias (optional).  N must be a multiple of 64.
// One wave computes a 16x64 output strip: A-frag loaded once per K step and
// reused across 4 B-frags -> 4 independent v_wmma chains per iteration.
// A-frag (16-bit 16x32): lanes<16 hold K {k..k+15} of row lane;
// lanes 16..31 hold K {k+16..k+31} of row lane-16.
// B-frag (32x16): lane holds 16 contiguous K of column lane%16 == 16
// contiguous bf16 of W's row (one 32B load).
// ---------------------------------------------------------------------------
__global__ __launch_bounds__(128)
void wmma_gemm_bf16(const bf16_t* __restrict__ A, const bf16_t* __restrict__ W,
                    const float* __restrict__ bias, float* __restrict__ C,
                    int M, int N, int K, int lda, int ldw, int ldc)
{
    const int lane = threadIdx.x;        // 0..31
    const int half = lane >> 4;          // 0/1
    const int l16  = lane & 15;
    const int tm = blockIdx.y * blockDim.y + threadIdx.y;   // M tile (16 rows)
    if (tm * 16 >= M) return;            // wave-uniform exit
    const int n0 = blockIdx.x * 64;      // 4 consecutive N tiles

    const bf16_t* arow = A + (size_t)(tm * 16 + l16) * lda;
    const bf16_t* w0   = W + (size_t)(n0 + l16) * ldw;

    v8f acc0 = {0.f,0.f,0.f,0.f,0.f,0.f,0.f,0.f};
    v8f acc1 = acc0, acc2 = acc0, acc3 = acc0;

#pragma unroll 2
    for (int k = 0; k < K; k += 32) {
        v8bf a0 = *(const v8bf*)(arow + k + half * 16);
        v8bf a1 = *(const v8bf*)(arow + k + half * 16 + 8);
        v16bf af;
#pragma unroll
        for (int i = 0; i < 8; ++i) { af[i] = a0[i]; af[i + 8] = a1[i]; }
        const int ko = k + half * 16;
        v16bf b0 = *(const v16bf*)(w0 + (size_t) 0 * ldw + ko);
        v16bf b1 = *(const v16bf*)(w0 + (size_t)16 * ldw + ko);
        v16bf b2 = *(const v16bf*)(w0 + (size_t)32 * ldw + ko);
        v16bf b3 = *(const v16bf*)(w0 + (size_t)48 * ldw + ko);
        acc0 = __builtin_amdgcn_wmma_f32_16x16x32_bf16(false, af, false, b0, (short)0, acc0, false, false);
        acc1 = __builtin_amdgcn_wmma_f32_16x16x32_bf16(false, af, false, b1, (short)0, acc1, false, false);
        acc2 = __builtin_amdgcn_wmma_f32_16x16x32_bf16(false, af, false, b2, (short)0, acc2, false, false);
        acc3 = __builtin_amdgcn_wmma_f32_16x16x32_bf16(false, af, false, b3, (short)0, acc3, false, false);
    }

    // D layout: element r -> row tm*16 + half*8 + r, col = tile_n*16 + l16
    v8f accs[4] = {acc0, acc1, acc2, acc3};
#pragma unroll
    for (int t = 0; t < 4; ++t) {
        const int col = n0 + t * 16 + l16;
        const float bv = bias ? bias[col] : 0.0f;
#pragma unroll
        for (int r = 0; r < 8; ++r) {
            const int row = tm * 16 + half * 8 + r;
            C[(size_t)row * ldc + col] = accs[t][r] + bv;
        }
    }
}

// ----------------------------- helpers -------------------------------------
__global__ void cvt_f32_bf16_kernel(const float* __restrict__ src,
                                    bf16_t* __restrict__ dst, int n)
{
    int i = blockIdx.x * blockDim.x + threadIdx.x;
    if (i < n) dst[i] = (bf16_t)src[i];
}

// dst[n, 0:Ka] = wa[n,:], dst[n, Ka:Ka+Kb] = wb[n,:]  (fp32 -> bf16)
__global__ void concat_cvt_kernel(const float* __restrict__ wa,
                                  const float* __restrict__ wb,
                                  bf16_t* __restrict__ dst,
                                  int N, int Ka, int Kb)
{
    int K = Ka + Kb;
    int i = blockIdx.x * blockDim.x + threadIdx.x;
    if (i >= N * K) return;
    int n = i / K, j = i % K;
    float v = (j < Ka) ? wa[(size_t)n * Ka + j] : wb[(size_t)n * Kb + (j - Ka)];
    dst[i] = (bf16_t)v;
}

// attn_w is H x (F+H); split into wF (H x F) and wH (H x H), both bf16.
__global__ void split_attn_w_kernel(const float* __restrict__ attn_w,
                                    bf16_t* __restrict__ wF, bf16_t* __restrict__ wH)
{
    int i = blockIdx.x * blockDim.x + threadIdx.x;
    if (i >= H_ * (F_ + H_)) return;
    int h = i / (F_ + H_), j = i % (F_ + H_);
    bf16_t v = (bf16_t)attn_w[i];
    if (j < F_) wF[h * F_ + j] = v;
    else        wH[h * H_ + (j - F_)] = v;
}

__global__ void bias_sum_kernel(const float* __restrict__ a, const float* __restrict__ b,
                                float* __restrict__ o, int n)
{
    int i = blockIdx.x * blockDim.x + threadIdx.x;
    if (i < n) o[i] = a[i] + b[i];
}

// zero fp32 cell state, zero h0 slot of xcat0, zero whole xcat1 ([h0|h1]).
__global__ void init_state_kernel(float* c0, float* c1, bf16_t* xcat0, bf16_t* xcat1)
{
    int i = blockIdx.x * blockDim.x + threadIdx.x;
    if (i < B_ * H_) {
        c0[i] = 0.f; c1[i] = 0.f;
        int b = i / H_, h = i % H_;
        xcat0[(size_t)b * (H_ + F_ + H_) + (H_ + F_) + h] = (bf16_t)0.f;
        xcat1[(size_t)b * (2 * H_) + h]      = (bf16_t)0.f;
        xcat1[(size_t)b * (2 * H_) + H_ + h] = (bf16_t)0.f;
    }
}

__global__ void zero_out_t0_kernel(float* __restrict__ out)
{
    int i = blockIdx.x * blockDim.x + threadIdx.x;   // over B_*V_
    if (i < B_ * V_) {
        int b = i / V_, v = i % V_;
        out[(size_t)b * T_ * V_ + v] = 0.f;
    }
}

// word fed at step s: START for s==0, else captions[:, s]. Writes embedding
// (bf16) into columns [0, H_) of xcat0 (row length H_+F_+H_).
__global__ void embed_gather_kernel(const float* __restrict__ emb,
                                    const long long* __restrict__ captions,
                                    bf16_t* __restrict__ xcat0, int s)
{
    int i = blockIdx.x * blockDim.x + threadIdx.x;   // over B_*H_
    if (i >= B_ * H_) return;
    int b = i / H_, h = i % H_;
    long long w = (s == 0) ? (long long)START_TOK : captions[(size_t)b * T_ + s];
    xcat0[(size_t)b * (H_ + F_ + H_) + h] = (bf16_t)emb[(size_t)w * H_ + h];
}

// logits[b,r] = attnv_b + sum_h attnv_w[h]*tanh(feat_proj[b,r,h]+hid_proj[b,h])
// one wave per (b,r); lanes stride over H then shuffle-reduce (wave32).
__global__ __launch_bounds__(256)
void attn_score_kernel(const float* __restrict__ feat_proj,
                       const float* __restrict__ hid_proj,
                       const float* __restrict__ attnv_w,
                       const float* __restrict__ attnv_b,
                       float* __restrict__ logits)
{
    int wave = (int)((blockIdx.x * blockDim.x + threadIdx.x) >> 5);
    int lane = threadIdx.x & 31;
    if (wave >= B_ * R_) return;
    int b = wave / R_;
    const float* fp = feat_proj + (size_t)wave * H_;
    const float* hp = hid_proj + (size_t)b * H_;
    float acc = 0.f;
    for (int h = lane; h < H_; h += 32)
        acc += attnv_w[h] * tanhf(fp[h] + hp[h]);
#pragma unroll
    for (int off = 16; off > 0; off >>= 1)
        acc += __shfl_xor(acc, off, 32);
    if (lane == 0) logits[wave] = acc + attnv_b[0];
}

// softmax over regions + context = sum_r w[b,r]*features[b,r,:]  (one block/b)
// context written (bf16) into columns [H_, H_+F_) of xcat0.
__global__ __launch_bounds__(256)
void softmax_context_kernel(const float* __restrict__ logits,
                            const float* __restrict__ features,
                            bf16_t* __restrict__ xcat0)
{
    __shared__ float sw[R_];
    __shared__ float red[256];
    int b = blockIdx.x;
    int tid = threadIdx.x;
    float v = (tid < R_) ? logits[b * R_ + tid] : -3.4e38f;
    red[tid] = v; __syncthreads();
    for (int off = 128; off > 0; off >>= 1) {
        if (tid < off) red[tid] = fmaxf(red[tid], red[tid + off]);
        __syncthreads();
    }
    float mx = red[0]; __syncthreads();
    float e = (tid < R_) ? __expf(v - mx) : 0.f;
    red[tid] = e; __syncthreads();
    for (int off = 128; off > 0; off >>= 1) {
        if (tid < off) red[tid] += red[tid + off];
        __syncthreads();
    }
    float inv = 1.0f / red[0];
    if (tid < R_) sw[tid] = e * inv;
    __syncthreads();
    for (int f = tid; f < F_; f += 256) {
        float acc = 0.f;
        const float* fb = features + (size_t)b * R_ * F_ + f;
        for (int r = 0; r < R_; ++r) acc += sw[r] * fb[(size_t)r * F_];
        xcat0[(size_t)b * (H_ + F_ + H_) + H_ + f] = (bf16_t)acc;
    }
}

// elementwise LSTM cell: gates (B x 4H, order i,f,g,o), updates c (fp32) and
// writes new hidden (bf16) into up to two consumer slots (strided rows).
__global__ void lstm_cell_kernel(const float* __restrict__ gates,
                                 float* __restrict__ c,
                                 bf16_t* __restrict__ hdst0, int stride0,
                                 bf16_t* __restrict__ hdst1, int stride1)
{
    int i = blockIdx.x * blockDim.x + threadIdx.x;   // over B_*H_
    if (i >= B_ * H_) return;
    int b = i / H_, h = i % H_;
    const float* g = gates + (size_t)b * 4 * H_;
    float gi = g[h], gf = g[H_ + h], gg = g[2 * H_ + h], go = g[3 * H_ + h];
    float si = 1.f / (1.f + __expf(-gi));
    float sf = 1.f / (1.f + __expf(-gf));
    float so = 1.f / (1.f + __expf(-go));
    float cn = sf * c[i] + si * tanhf(gg);
    c[i] = cn;
    bf16_t hv = (bf16_t)(so * tanhf(cn));
    hdst0[(size_t)b * stride0 + h] = hv;
    if (hdst1) hdst1[(size_t)b * stride1 + h] = hv;
}

// ----------------------------- launch --------------------------------------
static void gemm(const bf16_t* A, int lda, const bf16_t* W, int ldw,
                 const float* bias, float* C, int M, int N, int K, int ldc,
                 hipStream_t s)
{
    dim3 grid(N / 64, (M + 63) / 64);
    dim3 blk(32, 4);
    hipLaunchKernelGGL(wmma_gemm_bf16, grid, blk, 0, s,
                       A, W, bias, C, M, N, K, lda, ldw, ldc);
}

extern "C" void kernel_launch(void* const* d_in, const int* in_sizes, int n_in,
                              void* d_out, int out_size, void* d_ws, size_t ws_size,
                              hipStream_t stream)
{
    (void)in_sizes; (void)n_in; (void)out_size; (void)ws_size;
    const float*     features = (const float*)d_in[0];
    const long long* captions = (const long long*)d_in[1];
    const float*     emb      = (const float*)d_in[2];
    const float*     attn_w   = (const float*)d_in[3];
    const float*     attn_b   = (const float*)d_in[4];
    const float*     attnv_w  = (const float*)d_in[5];
    const float*     attnv_b  = (const float*)d_in[6];
    const float*     w_ih0    = (const float*)d_in[7];
    const float*     w_hh0    = (const float*)d_in[8];
    const float*     b_ih0    = (const float*)d_in[9];
    const float*     b_hh0    = (const float*)d_in[10];
    const float*     w_ih1    = (const float*)d_in[11];
    const float*     w_hh1    = (const float*)d_in[12];
    const float*     b_ih1    = (const float*)d_in[13];
    const float*     b_hh1    = (const float*)d_in[14];
    const float*     fc_w     = (const float*)d_in[15];
    const float*     fc_b     = (const float*)d_in[16];
    float* out = (float*)d_out;

    const int K0 = H_ + F_ + H_;   // 1536: [emb | ctx | h0]
    const int K1 = 2 * H_;         // 1024: [h0 | h1]

    char* p = (char*)d_ws;
    auto alloc = [&](size_t bytes) -> void* {
        uintptr_t q = ((uintptr_t)p + 255) & ~(uintptr_t)255;
        p = (char*)(q + bytes);
        return (void*)q;
    };
    bf16_t* attn_wF_b = (bf16_t*)alloc((size_t)H_ * F_ * 2);
    bf16_t* attn_wH_b = (bf16_t*)alloc((size_t)H_ * H_ * 2);
    bf16_t* wcat0     = (bf16_t*)alloc((size_t)4 * H_ * K0 * 2);   // [w_ih0 | w_hh0]
    bf16_t* wcat1     = (bf16_t*)alloc((size_t)4 * H_ * K1 * 2);   // [w_ih1 | w_hh1]
    bf16_t* fc_w_b    = (bf16_t*)alloc((size_t)V_ * H_ * 2);
    bf16_t* feats_b   = (bf16_t*)alloc((size_t)B_ * R_ * F_ * 2);
    float*  bsum0     = (float*)alloc((size_t)4 * H_ * 4);
    float*  bsum1     = (float*)alloc((size_t)4 * H_ * 4);
    float*  feat_proj = (float*)alloc((size_t)B_ * R_ * H_ * 4);
    float*  hid_proj  = (float*)alloc((size_t)B_ * H_ * 4);
    float*  logits    = (float*)alloc((size_t)B_ * R_ * 4);
    float*  gates     = (float*)alloc((size_t)B_ * 4 * H_ * 4);
    bf16_t* xcat0     = (bf16_t*)alloc((size_t)B_ * K0 * 2);       // [emb|ctx|h0]
    bf16_t* xcat1     = (bf16_t*)alloc((size_t)B_ * K1 * 2);       // [h0|h1]
    float*  c0        = (float*)alloc((size_t)B_ * H_ * 4);
    float*  c1        = (float*)alloc((size_t)B_ * H_ * 4);
    bf16_t* h1_view   = xcat1 + H_;   // h1 as strided view (lda = K1)

    const int TH = 256;
    auto blks = [](int n, int t) { return (n + t - 1) / t; };

    // ---- one-time: build bf16 weights ----
    concat_cvt_kernel<<<blks(4*H_*K0, TH), TH, 0, stream>>>(w_ih0, w_hh0, wcat0,
                                                            4*H_, H_ + F_, H_);
    concat_cvt_kernel<<<blks(4*H_*K1, TH), TH, 0, stream>>>(w_ih1, w_hh1, wcat1,
                                                            4*H_, H_, H_);
    cvt_f32_bf16_kernel<<<blks(V_*H_, TH), TH, 0, stream>>>(fc_w, fc_w_b, V_*H_);
    cvt_f32_bf16_kernel<<<blks(B_*R_*F_, TH), TH, 0, stream>>>(features, feats_b, B_*R_*F_);
    split_attn_w_kernel<<<blks(H_*(F_+H_), TH), TH, 0, stream>>>(attn_w, attn_wF_b, attn_wH_b);
    bias_sum_kernel<<<blks(4*H_, TH), TH, 0, stream>>>(b_ih0, b_hh0, bsum0, 4*H_);
    bias_sum_kernel<<<blks(4*H_, TH), TH, 0, stream>>>(b_ih1, b_hh1, bsum1, 4*H_);

    // ---- one-time: feat_proj = features @ attn_wF^T + attn_b (time-invariant) ----
    gemm(feats_b, F_, attn_wF_b, F_, attn_b, feat_proj, B_ * R_, H_, F_, H_, stream);

    // ---- init state + zero out[:,0,:] ----
    init_state_kernel<<<blks(B_*H_, TH), TH, 0, stream>>>(c0, c1, xcat0, xcat1);
    zero_out_t0_kernel<<<blks(B_*V_, TH), TH, 0, stream>>>(out);

    // ---- 29 sequential decode steps ----
    for (int s = 0; s < T_ - 1; ++s) {
        embed_gather_kernel<<<blks(B_*H_, TH), TH, 0, stream>>>(emb, captions, xcat0, s);

        // attention: hid_proj = h1 @ attn_wH^T ; score ; softmax+context
        gemm(h1_view, K1, attn_wH_b, H_, nullptr, hid_proj, B_, H_, H_, H_, stream);
        attn_score_kernel<<<(B_*R_)/8, TH, 0, stream>>>(feat_proj, hid_proj,
                                                        attnv_w, attnv_b, logits);
        softmax_context_kernel<<<B_, TH, 0, stream>>>(logits, features, xcat0);

        // LSTM layer 0 (fused): gates = [emb|ctx|h0] @ [w_ih0|w_hh0]^T + bsum0
        gemm(xcat0, K0, wcat0, K0, bsum0, gates, B_, 4*H_, K0, 4*H_, stream);
        lstm_cell_kernel<<<blks(B_*H_, TH), TH, 0, stream>>>(
            gates, c0, xcat0 + (H_ + F_), K0, xcat1, K1);   // h0 -> xcat0 & xcat1

        // LSTM layer 1 (fused): gates = [h0|h1] @ [w_ih1|w_hh1]^T + bsum1
        gemm(xcat1, K1, wcat1, K1, bsum1, gates, B_, 4*H_, K1, 4*H_, stream);
        lstm_cell_kernel<<<blks(B_*H_, TH), TH, 0, stream>>>(
            gates, c1, xcat1 + H_, K1, (bf16_t*)nullptr, 0); // h1 -> xcat1

        // vocab projection -> out[:, s+1, :]
        gemm(h1_view, K1, fc_w_b, H_, fc_b, out + (size_t)(s + 1) * V_,
             B_, V_, H_, T_ * V_, stream);
    }
}
